// RLFrameSelector_9225589752348
// MI455X (gfx1250) — matrix operand order repaired and checked by reading, not compiled, not run on hardware
//
#include <hip/hip_runtime.h>
#include <hip/hip_bf16.h>

// ---------------------------------------------------------------------------
// RLFrameSelector on gfx1250 (MI455X): bf16 WMMA MLP scoring + top-k + gather
// K-chunked double-buffered LDS staging; 2Mx4N register tile per wave so each
// B fragment feeds two WMMAs (1.5 ds_load_b128 per WMMA).
// ---------------------------------------------------------------------------

typedef __attribute__((ext_vector_type(16))) __bf16          v16bf;
typedef __attribute__((ext_vector_type(16))) unsigned short  v16us;
typedef __attribute__((ext_vector_type(8)))  float           v8f;

#define B_DIM   32
#define T_DIM   2048
#define F_DIM   512
#define U_DIM   128
#define K_SEL   64
#define ROWS    (B_DIM * T_DIM)     // 65536 frames
#define TILE_M  128                 // rows per block in score kernel
#define KC      128                 // K-chunk width
#define NCHUNK  (F_DIM / KC)        // 4 chunks
#define PC      136                 // LDS row pitch (ushorts): 128 + 8 pad, 272B rows (16B-aligned)
#define CHUNK_WORDS (TILE_M * PC)   // per-buffer ushort count (x and W1t identical shape)
#define NEG_BIG (-1.0e9f)
#define SENT    (-3.402823466e38f)

union FragU {
    v16us us;
    v16bf bf;
    uint4 q[2];
};

__device__ __forceinline__ unsigned short f2bf(float f) {
    unsigned int u = __float_as_uint(f);
    u += 0x7FFFu + ((u >> 16) & 1u);   // round-to-nearest-even
    return (unsigned short)(u >> 16);
}

// ---------------------------------------------------------------------------
// Kernel 1: scores[row] = relu(x[row,:] @ W1 + b1) @ W2 + b2  (masked)
// 512 blocks x 256 threads (8 waves). 139KB dynamic LDS -> 2 blocks/WGP.
// Wave tile: 32 rows x 64 cols = acc[2][4]; waves (w>>2) split the N range.
// ---------------------------------------------------------------------------
__global__ void rlfs_score_kernel(const float* __restrict__ x,
                                  const float* __restrict__ W1,
                                  const float* __restrict__ b1,
                                  const float* __restrict__ W2,
                                  const float* __restrict__ b2,
                                  float* __restrict__ scores)
{
    extern __shared__ unsigned char smem_raw[];
    // double-buffered chunks: xs[2][128][PC], w1t[2][128][PC]
    unsigned short* xs0  = (unsigned short*)smem_raw;
    unsigned short* xs1  = xs0 + CHUNK_WORDS;
    unsigned short* w1t0 = xs1 + CHUNK_WORDS;
    unsigned short* w1t1 = w1t0 + CHUNK_WORDS;
    __shared__ int   valid[TILE_M];
    __shared__ float psum[2][TILE_M];

    const int tid  = threadIdx.x;
    const int row0 = blockIdx.x * TILE_M;

    if (tid < TILE_M) valid[tid] = 0;

    const int wave = tid >> 5;
    const int lane = tid & 31;
    const int half = lane >> 4;        // K-group selector per ISA wave32 layout
    const int ln   = lane & 15;
    const int mg   = (wave & 3) * 32;  // wave's local row base (two 16-row tiles)
    const int nhq  = wave >> 2;        // which N-half (0: cols 0-63, 1: cols 64-127)
    const int nh   = nhq * 64;

    const float4* xg = (const float4*)x;

    v8f acc0[4], acc1[4];
    #pragma unroll
    for (int t = 0; t < 4; ++t) {
        acc0[t] = (v8f){0.f,0.f,0.f,0.f,0.f,0.f,0.f,0.f};
        acc1[t] = (v8f){0.f,0.f,0.f,0.f,0.f,0.f,0.f,0.f};
    }

    __syncthreads();   // valid[] initialized

    // ---- stage chunk 0 ----
    {
        unsigned short* xs = xs0;  unsigned short* wt = w1t0;
        #pragma unroll 4
        for (int i = tid; i < TILE_M * (KC / 4); i += 256) {   // 4096 float4
            const int r  = i >> 5;           // 32 float4 per chunk-row
            const int c4 = i & 31;
            float4 v = xg[(size_t)(row0 + r) * (F_DIM / 4) + c4];
            unsigned short* d = xs + r * PC + (c4 << 2);
            d[0] = f2bf(v.x); d[1] = f2bf(v.y); d[2] = f2bf(v.z); d[3] = f2bf(v.w);
            if (v.x != 0.f || v.y != 0.f || v.z != 0.f || v.w != 0.f) valid[r] = 1;
        }
        #pragma unroll 4
        for (int i = tid; i < KC * U_DIM; i += 256) {          // 16384 floats
            const int kk = i >> 7;
            const int n  = i & 127;
            wt[n * PC + kk] = f2bf(W1[(size_t)kk * U_DIM + n]);
        }
    }
    __syncthreads();

    // ---- pipelined chunk loop: load c+1 while computing c ----
    for (int c = 0; c < NCHUNK; ++c) {
        unsigned short* xsC = (c & 1) ? xs1 : xs0;
        unsigned short* wtC = (c & 1) ? w1t1 : w1t0;

        if (c + 1 < NCHUNK) {
            unsigned short* xsN = ((c + 1) & 1) ? xs1 : xs0;
            unsigned short* wtN = ((c + 1) & 1) ? w1t1 : w1t0;
            const int kc0 = (c + 1) * KC;
            #pragma unroll 4
            for (int i = tid; i < TILE_M * (KC / 4); i += 256) {
                const int r  = i >> 5;
                const int c4 = i & 31;
                float4 v = xg[(size_t)(row0 + r) * (F_DIM / 4) + (kc0 >> 2) + c4];
                unsigned short* d = xsN + r * PC + (c4 << 2);
                d[0] = f2bf(v.x); d[1] = f2bf(v.y); d[2] = f2bf(v.z); d[3] = f2bf(v.w);
                if (v.x != 0.f || v.y != 0.f || v.z != 0.f || v.w != 0.f) valid[r] = 1;
            }
            #pragma unroll 4
            for (int i = tid; i < KC * U_DIM; i += 256) {
                const int kk = i >> 7;
                const int n  = i & 127;
                wtN[n * PC + kk] = f2bf(W1[(size_t)(kc0 + kk) * U_DIM + n]);
            }
        }

        // ---- WMMA over this chunk: 4 K-steps; A pair + 4 shared B frags ----
        const unsigned short* arow0 = xsC + (mg + ln) * PC;
        const unsigned short* arow1 = xsC + (mg + 16 + ln) * PC;
        #pragma unroll
        for (int kb = 0; kb < KC; kb += 32) {
            FragU a0, a1;
            {
                const uint4* ap0 = (const uint4*)(arow0 + kb + half * 8);
                a0.q[0] = ap0[0];        // K = kb + 8*half + [0..7]
                a0.q[1] = ap0[2];        // K = kb + 16 + 8*half + [0..7]
                const uint4* ap1 = (const uint4*)(arow1 + kb + half * 8);
                a1.q[0] = ap1[0];
                a1.q[1] = ap1[2];
            }
            FragU bfr[4];
            #pragma unroll
            for (int t = 0; t < 4; ++t) {
                const uint4* bp = (const uint4*)(wtC + (nh + t * 16 + ln) * PC + kb + half * 8);
                bfr[t].q[0] = bp[0];
                bfr[t].q[1] = bp[2];
            }
            #pragma unroll
            for (int t = 0; t < 4; ++t) {
                acc0[t] = __builtin_amdgcn_wmma_f32_16x16x32_bf16(
                    false, a0.bf, false, bfr[t].bf, (short)0, acc0[t], false, false);
                acc1[t] = __builtin_amdgcn_wmma_f32_16x16x32_bf16(
                    false, a1.bf, false, bfr[t].bf, (short)0, acc1[t], false, false);
            }
        }
        __syncthreads();   // next-chunk staging complete / buffer safe to overwrite
    }

    // ---- epilogue: bias + relu + dot(W2) over this wave's N-half ----
    float p0[8] = {0,0,0,0,0,0,0,0};
    float p1[8] = {0,0,0,0,0,0,0,0};
    #pragma unroll
    for (int t = 0; t < 4; ++t) {
        const int n      = nh + t * 16 + ln;
        const float bias = b1[n];
        const float w2v  = W2[n];
        #pragma unroll
        for (int v = 0; v < 8; ++v) {
            float h0 = acc0[t][v] + bias;
            float h1 = acc1[t][v] + bias;
            h0 = h0 > 0.f ? h0 : 0.f;
            h1 = h1 > 0.f ? h1 : 0.f;
            p0[v] += h0 * w2v;
            p1[v] += h1 * w2v;
        }
    }
    #pragma unroll
    for (int v = 0; v < 8; ++v) {
        float q0 = p0[v], q1 = p1[v];
        q0 += __shfl_xor(q0, 1, 32);  q1 += __shfl_xor(q1, 1, 32);
        q0 += __shfl_xor(q0, 2, 32);  q1 += __shfl_xor(q1, 2, 32);
        q0 += __shfl_xor(q0, 4, 32);  q1 += __shfl_xor(q1, 4, 32);
        q0 += __shfl_xor(q0, 8, 32);  q1 += __shfl_xor(q1, 8, 32);
        if (ln == 0) {
            const int m0 = mg + v + 8 * half;        // C/D layout: M = v + 8*half
            psum[nhq][m0]      = q0;
            psum[nhq][m0 + 16] = q1;
        }
    }
    __syncthreads();

    if (tid < TILE_M) {
        float sc = psum[0][tid] + psum[1][tid] + b2[0];
        if (!valid[tid]) sc = NEG_BIG;
        scores[row0 + tid] = sc;
    }
}

// ---------------------------------------------------------------------------
// Kernel 2: per batch row, select top-64 of 2048 (stable: ties -> lower index)
// ---------------------------------------------------------------------------
__global__ void rlfs_topk_kernel(const float* __restrict__ scores,
                                 float* __restrict__ selmask)
{
    __shared__ float sc[T_DIM];
    __shared__ float redv[256];
    __shared__ int   redi[256];

    const int b   = blockIdx.x;
    const int tid = threadIdx.x;

    for (int i = tid; i < T_DIM; i += 256) sc[i] = scores[b * T_DIM + i];
    __syncthreads();

    for (int iter = 0; iter < K_SEL; ++iter) {
        float bv = SENT;
        int   bi = 0x7fffffff;
        for (int j = tid; j < T_DIM; j += 256) {
            const float v = sc[j];
            if (v > bv) { bv = v; bi = j; }
        }
        redv[tid] = bv; redi[tid] = bi;
        __syncthreads();
        for (int s = 128; s > 0; s >>= 1) {
            if (tid < s) {
                const float ov = redv[tid + s];
                const int   oi = redi[tid + s];
                if (ov > redv[tid] || (ov == redv[tid] && oi < redi[tid])) {
                    redv[tid] = ov; redi[tid] = oi;
                }
            }
            __syncthreads();
        }
        if (tid == 0) sc[redi[0]] = SENT;   // mark selected
        __syncthreads();
    }

    for (int i = tid; i < T_DIM; i += 256)
        selmask[b * T_DIM + i] = (sc[i] == SENT) ? 1.0f : 0.0f;
}

// ---------------------------------------------------------------------------
// Kernel 3: out = x * sel  — skip reading x for dropped frames (97% of them)
// ---------------------------------------------------------------------------
__global__ void rlfs_apply_kernel(const float4* __restrict__ x4,
                                  const float* __restrict__ selmask,
                                  float4* __restrict__ out4)
{
    const int idx   = blockIdx.x * 256 + threadIdx.x;   // over ROWS*F/4 float4s
    const int frame = idx >> 7;                         // 128 float4 per frame
    float4 r = make_float4(0.f, 0.f, 0.f, 0.f);
    if (selmask[frame] != 0.0f) r = x4[idx];
    out4[idx] = r;
}

// ---------------------------------------------------------------------------
extern "C" void kernel_launch(void* const* d_in, const int* in_sizes, int n_in,
                              void* d_out, int out_size, void* d_ws, size_t ws_size,
                              hipStream_t stream)
{
    const float* x  = (const float*)d_in[0];   // [32,2048,512]
    const float* W1 = (const float*)d_in[1];   // [512,128]
    const float* b1 = (const float*)d_in[2];   // [128]
    const float* W2 = (const float*)d_in[3];   // [128,1]
    const float* b2 = (const float*)d_in[4];   // [1]
    // d_in[5] = k (fixed 64)
    (void)in_sizes; (void)n_in; (void)out_size; (void)ws_size;

    float* scores  = (float*)d_ws;             // 65536 floats
    float* selmask = scores + ROWS;            // 65536 floats

    const size_t dyn_lds = (size_t)4 * CHUNK_WORDS * sizeof(unsigned short); // 139264 B

    rlfs_score_kernel<<<ROWS / TILE_M, 256, dyn_lds, stream>>>(x, W1, b1, W2, b2, scores);
    rlfs_topk_kernel<<<B_DIM, 256, 0, stream>>>(scores, selmask);
    rlfs_apply_kernel<<<(ROWS * (F_DIM / 4)) / 256, 256, 0, stream>>>(
        (const float4*)x, selmask, (float4*)d_out);
}
